// PNVAE_48361331753187
// MI455X (gfx1250) — matrix operand order repaired and checked by reading, not compiled, not run on hardware
//
#include <hip/hip_runtime.h>

// ---------------- problem constants ----------------
#define N_B     128
#define P_PTS   100
#define NP      (N_B * P_PTS)      // 12800
#define KNN     16
#define M_EDGE  (NP * KNN)         // 204800
#define LATENT  16
#define BN_EPS  1e-3f

typedef __attribute__((ext_vector_type(16))) _Float16 v16h;
typedef __attribute__((ext_vector_type(8)))  _Float16 v8h;
typedef __attribute__((ext_vector_type(8)))  float    v8f;

// A-fragment K base per lane (ISA 7.12.2, f16 16x32 A):
// lane&15 = M row; halves 0..7 -> K = 8*(lane>>4)+h ; halves 8..15 -> +16
// B 32x16 f16: lane&15 = N col; half h -> K = 8*(lane>>4)+(h&7)+16*(h>>3)
// C/D 16x16 f32: lane&15 = N; vgpr r -> M = r + 8*(lane>>4)

// =====================================================================
// kNN top-(k+1) selection (strict-< keeps earliest index on ties), drop self
// =====================================================================
template<int CD>
__global__ void knn_topk_kernel(const float* __restrict__ pts, int* __restrict__ idx_out) {
  int t = blockIdx.x * blockDim.x + threadIdx.x;
  if (t >= NP) return;
  int n = t / P_PTS, p = t % P_PTS;
  const float* bp = pts + (size_t)n * P_PTS * CD;
  float dv[KNN + 1];
  int   di[KNN + 1];
#pragma unroll
  for (int i = 0; i <= KNN; ++i) { dv[i] = 3.0e38f; di[i] = 0; }
  for (int q = 0; q < P_PTS; ++q) {
    float d = 0.f;
    for (int c = 0; c < CD; ++c) {
      float df = bp[(size_t)p * CD + c] - bp[(size_t)q * CD + c];
      d += df * df;
    }
    if (d < dv[KNN]) {
      int i = KNN;
      while (i > 0 && d < dv[i - 1]) { dv[i] = dv[i - 1]; di[i] = di[i - 1]; --i; }
      dv[i] = d; di[i] = q;
    }
  }
  for (int j = 0; j < KNN; ++j) idx_out[(size_t)t * KNN + j] = di[j + 1];
}

// =====================================================================
// BN over last-dim channels (input feature pre-norm)
// =====================================================================
__global__ void bn_fts_kernel(const float* __restrict__ x,
                              const float* __restrict__ g, const float* __restrict__ b,
                              const float* __restrict__ m, const float* __restrict__ v,
                              float* __restrict__ y, int total, int C) {
  int t = blockIdx.x * blockDim.x + threadIdx.x;
  if (t >= total) return;
  int c = t % C;
  y[t] = (x[t] - m[c]) * g[c] * rsqrtf(v[c] + BN_EPS) + b[c];
}

// =====================================================================
// Pack weights (KxC f32, zero-padded to KPAD) into B-fragment order:
// Wp[(((ntile*NC)+kc)*32 + lane)*16 + h] = W[(kc*32+(lane>>4)*8+(h&7)+(h>>3)*16)*C + ntile*16+(lane&15)]
// =====================================================================
__global__ void pack_w_kernel(const float* __restrict__ W, _Float16* __restrict__ Wp,
                              int K, int KPAD, int C) {
  int t = blockIdx.x * blockDim.x + threadIdx.x;
  if (t >= KPAD * C) return;
  int h    = t & 15;
  int lane = (t >> 4) & 31;
  int rest = t >> 9;
  int nc   = KPAD / 32;
  int kc    = rest % nc;
  int ntile = rest / nc;
  int col = ntile * 16 + (lane & 15);
  int k   = kc * 32 + (lane >> 4) * 8 + (h & 7) + (h >> 3) * 16;
  Wp[t] = (_Float16)((k < K) ? W[(size_t)k * C + col] : 0.f);
}

__device__ __forceinline__ v16h cvt_frag(v8f lo, v8f hi) {
  v16h a;
#pragma unroll
  for (int h = 0; h < 8; ++h) { a[h] = (_Float16)lo[h]; a[h + 8] = (_Float16)hi[h]; }
  return a;
}

// =====================================================================
// GEMM #1: first conv of a block; A = virtual edge tensor (center || nbr-center)
// gathered on the fly. out = relu(bn(A@W)) as f16. One wave -> 16x64 strip.
// =====================================================================
template<int CIN, int C>
__global__ void gemm_edge_wmma(const float* __restrict__ fts, const int* __restrict__ idx,
                               const _Float16* __restrict__ Wp,
                               const float* __restrict__ gg, const float* __restrict__ gb,
                               const float* __restrict__ gm, const float* __restrict__ gv,
                               _Float16* __restrict__ out) {
  constexpr int K    = 2 * CIN;
  constexpr int KPAD = (K + 31) & ~31;
  constexpr int NC   = KPAD / 32;
  constexpr int NT   = 4;
  constexpr int MT   = M_EDGE / 16;
  constexpr int NG   = C / (16 * NT);
  int lane = threadIdx.x & 31;
  int wave = threadIdx.x >> 5;
  int gt = blockIdx.x * (blockDim.x >> 5) + wave;
  if (gt >= MT * NG) return;                 // wave-uniform
  int mtile = gt % MT;
  int ngrp  = gt / MT;

  int arow = mtile * 16 + (lane & 15);
  int np_i = arow >> 4;
  int j    = arow & 15;
  int n    = np_i / P_PTS;
  int q    = idx[(size_t)np_i * KNN + j];
  const float* fc = fts + (size_t)np_i * CIN;
  const float* fq = fts + (size_t)(n * P_PTS + q) * CIN;
  int akb = (lane >> 4) * 8;

  v8f acc[NT] = {};
#pragma unroll
  for (int kc = 0; kc < NC; ++kc) {
    v16h a;
    if constexpr (CIN == 3) {
      float c0 = fc[0], c1 = fc[1], c2 = fc[2];
      float q0 = fq[0], q1 = fq[1], q2 = fq[2];
      bool low = (lane < 16);
      const _Float16 z = (_Float16)0.f;
#pragma unroll
      for (int h = 0; h < 16; ++h) a[h] = z;
      a[0] = low ? (_Float16)c0 : z;
      a[1] = low ? (_Float16)c1 : z;
      a[2] = low ? (_Float16)c2 : z;
      a[3] = low ? (_Float16)(q0 - c0) : z;
      a[4] = low ? (_Float16)(q1 - c1) : z;
      a[5] = low ? (_Float16)(q2 - c2) : z;
    } else {
      int kk = kc * 32;
      v8f lo, hi;
      if (kk + 31 < CIN) {                 // folds after unroll: pure center chunk
        lo = *(const v8f*)(fc + kk + akb);
        hi = *(const v8f*)(fc + kk + 16 + akb);
      } else {                             // pure (neighbor - center) chunk
        int kd = kk - CIN;
        v8f lq = *(const v8f*)(fq + kd + akb);
        v8f lc = *(const v8f*)(fc + kd + akb);
        v8f hq = *(const v8f*)(fq + kd + 16 + akb);
        v8f hc = *(const v8f*)(fc + kd + 16 + akb);
        lo = lq - lc;
        hi = hq - hc;
      }
      a = cvt_frag(lo, hi);
    }
#pragma unroll
    for (int t = 0; t < NT; ++t) {
      v16h b = *(const v16h*)(Wp + (((size_t)(ngrp * NT + t) * NC + kc) * 32 + lane) * 16);
      acc[t] = __builtin_amdgcn_wmma_f32_16x16x32_f16(false, a, false, b,
                                                      (short)0, acc[t], false, false);
    }
  }
  int mbase = mtile * 16 + (lane >> 4) * 8;
#pragma unroll
  for (int t = 0; t < NT; ++t) {
    int bcol = (ngrp * NT + t) * 16 + (lane & 15);
    float sg = gg[bcol], sb = gb[bcol], sm = gm[bcol];
    float si = rsqrtf(gv[bcol] + BN_EPS);
#pragma unroll
    for (int r = 0; r < 8; ++r) {
      float val = fmaxf((acc[t][r] - sm) * sg * si + sb, 0.f);
      out[(size_t)(mbase + r) * C + bcol] = (_Float16)val;
    }
  }
}

// =====================================================================
// GEMM #2: f16 activations in, relu(bn(A@W)) f16 out (convs 2 & 3)
// =====================================================================
template<int K, int C>
__global__ void gemm_f16_wmma(const _Float16* __restrict__ A, const _Float16* __restrict__ Wp,
                              const float* __restrict__ gg, const float* __restrict__ gb,
                              const float* __restrict__ gm, const float* __restrict__ gv,
                              _Float16* __restrict__ out) {
  constexpr int NC = K / 32;
  constexpr int NT = 4;
  constexpr int MT = M_EDGE / 16;
  constexpr int NG = C / (16 * NT);
  int lane = threadIdx.x & 31;
  int wave = threadIdx.x >> 5;
  int gt = blockIdx.x * (blockDim.x >> 5) + wave;
  if (gt >= MT * NG) return;
  int mtile = gt % MT;
  int ngrp  = gt / MT;

  const _Float16* arp = A + (size_t)(mtile * 16 + (lane & 15)) * K;
  int akb = (lane >> 4) * 8;

  v8f acc[NT] = {};
#pragma unroll
  for (int kc = 0; kc < NC; ++kc) {
    v8h lo = *(const v8h*)(arp + kc * 32 + akb);
    v8h hi = *(const v8h*)(arp + kc * 32 + 16 + akb);
    v16h a = __builtin_shufflevector(lo, hi, 0, 1, 2, 3, 4, 5, 6, 7,
                                     8, 9, 10, 11, 12, 13, 14, 15);
#pragma unroll
    for (int t = 0; t < NT; ++t) {
      v16h b = *(const v16h*)(Wp + (((size_t)(ngrp * NT + t) * NC + kc) * 32 + lane) * 16);
      acc[t] = __builtin_amdgcn_wmma_f32_16x16x32_f16(false, a, false, b,
                                                      (short)0, acc[t], false, false);
    }
  }
  int mbase = mtile * 16 + (lane >> 4) * 8;
#pragma unroll
  for (int t = 0; t < NT; ++t) {
    int bcol = (ngrp * NT + t) * 16 + (lane & 15);
    float sg = gg[bcol], sb = gb[bcol], sm = gm[bcol];
    float si = rsqrtf(gv[bcol] + BN_EPS);
#pragma unroll
    for (int r = 0; r < 8; ++r) {
      float val = fmaxf((acc[t][r] - sm) * sg * si + sb, 0.f);
      out[(size_t)(mbase + r) * C + bcol] = (_Float16)val;
    }
  }
}

// =====================================================================
// GEMM #3: f32 A, f32 out. epilogue: (+bias) -> bn -> (+add) -> relu
// shortcut (+agg), dec0 (+bias), decoder convs.
// =====================================================================
template<int K, int C, bool BIAS, bool ADD>
__global__ void gemm_f32_wmma(const float* __restrict__ A, const _Float16* __restrict__ Wp,
                              const float* __restrict__ bias,
                              const float* __restrict__ gg, const float* __restrict__ gb,
                              const float* __restrict__ gm, const float* __restrict__ gv,
                              const float* __restrict__ add,
                              float* __restrict__ out, int M) {
  constexpr int KPAD = (K + 31) & ~31;
  constexpr int NC   = KPAD / 32;
  constexpr int NT   = 4;
  constexpr int NG   = C / (16 * NT);
  int lane = threadIdx.x & 31;
  int wave = threadIdx.x >> 5;
  int mt = M / 16;
  int gt = blockIdx.x * (blockDim.x >> 5) + wave;
  if (gt >= mt * NG) return;
  int mtile = gt % mt;
  int ngrp  = gt / mt;

  const float* arp = A + (size_t)(mtile * 16 + (lane & 15)) * K;
  int akb = (lane >> 4) * 8;

  v8f acc[NT] = {};
#pragma unroll
  for (int kc = 0; kc < NC; ++kc) {
    v16h a;
    if constexpr (K % 32 == 0) {
      v8f lo = *(const v8f*)(arp + kc * 32 + akb);
      v8f hi = *(const v8f*)(arp + kc * 32 + 16 + akb);
      a = cvt_frag(lo, hi);
    } else if constexpr (K == 16) {
      v8f lo = *(const v8f*)(arp + akb);   // lanes 0-15: K0..7; lanes 16-31: K8..15
#pragma unroll
      for (int h = 0; h < 8; ++h) { a[h] = (_Float16)lo[h]; a[h + 8] = (_Float16)0.f; }
    } else {                               // K == 3 (block-0 shortcut)
      float t0 = arp[0], t1 = arp[1], t2 = arp[2];
      bool low = (lane < 16);
      const _Float16 z = (_Float16)0.f;
#pragma unroll
      for (int h = 0; h < 16; ++h) a[h] = z;
      a[0] = low ? (_Float16)t0 : z;
      a[1] = low ? (_Float16)t1 : z;
      a[2] = low ? (_Float16)t2 : z;
    }
#pragma unroll
    for (int t = 0; t < NT; ++t) {
      v16h b = *(const v16h*)(Wp + (((size_t)(ngrp * NT + t) * NC + kc) * 32 + lane) * 16);
      acc[t] = __builtin_amdgcn_wmma_f32_16x16x32_f16(false, a, false, b,
                                                      (short)0, acc[t], false, false);
    }
  }
  int mbase = mtile * 16 + (lane >> 4) * 8;
#pragma unroll
  for (int t = 0; t < NT; ++t) {
    int bcol = (ngrp * NT + t) * 16 + (lane & 15);
    float bi = 0.f;
    if constexpr (BIAS) bi = bias[bcol];
    float sg = gg[bcol], sb = gb[bcol], sm = gm[bcol];
    float si = rsqrtf(gv[bcol] + BN_EPS);
#pragma unroll
    for (int r = 0; r < 8; ++r) {
      float val = (acc[t][r] + bi - sm) * sg * si + sb;
      if constexpr (ADD) val += add[(size_t)(mbase + r) * C + bcol];
      out[(size_t)(mbase + r) * C + bcol] = fmaxf(val, 0.f);
    }
  }
}

// =====================================================================
// max over the neighbor axis: (NP*KNN, C) f16 -> (NP, C) f32
// =====================================================================
__global__ void maxk_kernel(const _Float16* __restrict__ in, float* __restrict__ agg, int C) {
  int t = blockIdx.x * blockDim.x + threadIdx.x;
  if (t >= NP * C) return;
  int np = t / C, c = t % C;
  float mx = -3.0e38f;
#pragma unroll
  for (int j = 0; j < KNN; ++j)
    mx = fmaxf(mx, (float)in[((size_t)np * KNN + j) * C + c]);
  agg[t] = mx;
}

// mean pool over points: (N, P, C) -> (N, C)
__global__ void pool_kernel(const float* __restrict__ fts, float* __restrict__ pool, int C) {
  int t = blockIdx.x * blockDim.x + threadIdx.x;
  if (t >= N_B * C) return;
  int nb = t / C, c = t % C;
  float s = 0.f;
  for (int p = 0; p < P_PTS; ++p) s += fts[((size_t)nb * P_PTS + p) * C + c];
  pool[t] = s * (1.0f / P_PTS);
}

// VAE head
__global__ void encoder_kernel(const float* __restrict__ pool,
                               const float* __restrict__ Wm, const float* __restrict__ Wv,
                               const float* __restrict__ bm, const float* __restrict__ bv,
                               const float* __restrict__ eps,
                               float* __restrict__ z, float* __restrict__ zm_o,
                               float* __restrict__ zv_o, int Cin) {
  int t = blockIdx.x * blockDim.x + threadIdx.x;
  if (t >= N_B * LATENT) return;
  int nb = t / LATENT, l = t % LATENT;
  float am = bm[l], av = bv[l];
  for (int c = 0; c < Cin; ++c) {
    float pv = pool[(size_t)nb * Cin + c];
    am += pv * Wm[(size_t)c * LATENT + l];
    av += pv * Wv[(size_t)c * LATENT + l];
  }
  am = fmaxf(am, 0.f);
  av = fmaxf(av, 0.f);
  zm_o[t] = am;
  zv_o[t] = av;
  z[t] = am + __expf(0.5f * av) * eps[t];
}

// decoder output head: (NP,64)@(64,3)+b, relu
__global__ void dec_out_kernel(const float* __restrict__ x, const float* __restrict__ Wout,
                               const float* __restrict__ bout, float* __restrict__ out) {
  int t = blockIdx.x * blockDim.x + threadIdx.x;
  if (t >= NP * 3) return;
  int np = t / 3, o = t % 3;
  float s = bout[o];
#pragma unroll
  for (int c = 0; c < 64; ++c) s += x[(size_t)np * 64 + c] * Wout[c * 3 + o];
  out[t] = fmaxf(s, 0.f);
}

// =====================================================================
// host orchestration
// =====================================================================
static inline int gemm_grid(int M, int C) { return ((M / 16) * (C / 64) + 7) / 8; }
static inline int pack_grid(int KPAD, int C) { return (KPAD * C + 255) / 256; }

extern "C" void kernel_launch(void* const* d_in, const int* in_sizes, int n_in,
                              void* d_out, int out_size, void* d_ws, size_t ws_size,
                              hipStream_t stream) {
  (void)in_sizes; (void)n_in; (void)out_size; (void)ws_size;
  const float* points   = (const float*)d_in[0];
  const float* features = (const float*)d_in[1];
  const float* eps      = (const float*)d_in[2];
  void* const* prm = const_cast<void* const*>(d_in + 3);
  auto PW = [&](int i) { return (const float*)prm[i]; };

  // jax tree-flatten order (sorted dict keys; bn -> b,g,m,v):
  // blocks[b]: conv c -> W(+0) b(+1) g(+2) m(+3) v(+4); sc_W(+15) sc_bn(+16..19)
  const int I_DEC_W0 = 60, I_DEC_WOUT = 61, I_DEC_B0 = 62, I_DEC_BN0 = 63,
            I_DEC_BOUT = 67, I_DEC_CONV = 68, I_ENC_WM = 78, I_ENC_WV = 79,
            I_ENC_BM = 80, I_ENC_BV = 81, I_FTS_BN = 82;

  // workspace carve-out
  char* ws = (char*)d_ws;
  size_t off = 0;
  auto carve = [&](size_t bytes) {
    void* p = ws + off;
    off = (off + bytes + 255) & ~(size_t)255;
    return p;
  };
  float*    ftsA  = (float*)carve((size_t)NP * 256 * sizeof(float));
  float*    ftsB  = (float*)carve((size_t)NP * 256 * sizeof(float));
  int*      idx   = (int*)carve((size_t)NP * KNN * sizeof(int));
  _Float16* cvA   = (_Float16*)carve((size_t)M_EDGE * 256 * sizeof(_Float16));
  _Float16* cvB   = (_Float16*)carve((size_t)M_EDGE * 256 * sizeof(_Float16));
  float*    agg   = (float*)carve((size_t)NP * 256 * sizeof(float));
  float*    poolb = (float*)carve((size_t)N_B * 256 * sizeof(float));
  float*    dxA   = (float*)carve((size_t)NP * 128 * sizeof(float));
  float*    dxB   = (float*)carve((size_t)NP * 128 * sizeof(float));
  _Float16* wpk   = (_Float16*)carve((size_t)32 * 6400 * sizeof(_Float16)); // max KPAD*C

  // outputs: [z | z_mean | z_log_var | dec]
  float* z   = (float*)d_out;
  float* zm  = z + N_B * LATENT;
  float* zv  = zm + N_B * LATENT;
  float* dec = zv + N_B * LATENT;

  // features BN
  bn_fts_kernel<<<(NP * 3 + 255) / 256, 256, 0, stream>>>(
      features, PW(I_FTS_BN + 1), PW(I_FTS_BN + 0), PW(I_FTS_BN + 2), PW(I_FTS_BN + 3),
      ftsA, NP * 3, 3);

  float* fcur = ftsA;
  float* fnxt = ftsB;

  // ---------------- block 0: cin=3, ch=64 ----------------
  {
    const int base = 0;
    knn_topk_kernel<2><<<(NP + 63) / 64, 64, 0, stream>>>(points, idx);

    pack_w_kernel<<<pack_grid(32, 64), 256, 0, stream>>>(PW(base + 0), wpk, 6, 32, 64);
    gemm_edge_wmma<3, 64><<<gemm_grid(M_EDGE, 64), 256, 0, stream>>>(
        fcur, idx, wpk, PW(base + 2), PW(base + 1), PW(base + 3), PW(base + 4), cvA);

    pack_w_kernel<<<pack_grid(64, 64), 256, 0, stream>>>(PW(base + 5), wpk, 64, 64, 64);
    gemm_f16_wmma<64, 64><<<gemm_grid(M_EDGE, 64), 256, 0, stream>>>(
        cvA, wpk, PW(base + 7), PW(base + 6), PW(base + 8), PW(base + 9), cvB);

    pack_w_kernel<<<pack_grid(64, 64), 256, 0, stream>>>(PW(base + 10), wpk, 64, 64, 64);
    gemm_f16_wmma<64, 64><<<gemm_grid(M_EDGE, 64), 256, 0, stream>>>(
        cvB, wpk, PW(base + 12), PW(base + 11), PW(base + 13), PW(base + 14), cvA);

    maxk_kernel<<<(NP * 64 + 255) / 256, 256, 0, stream>>>(cvA, agg, 64);

    pack_w_kernel<<<pack_grid(32, 64), 256, 0, stream>>>(PW(base + 15), wpk, 3, 32, 64);
    gemm_f32_wmma<3, 64, false, true><<<gemm_grid(NP, 64), 256, 0, stream>>>(
        fcur, wpk, nullptr, PW(base + 17), PW(base + 16), PW(base + 18), PW(base + 19),
        agg, fnxt, NP);
    float* tmp = fcur; fcur = fnxt; fnxt = tmp;
  }

  // ---------------- block 1: cin=64, ch=128 ----------------
  {
    const int base = 20;
    knn_topk_kernel<64><<<(NP + 63) / 64, 64, 0, stream>>>(fcur, idx);

    pack_w_kernel<<<pack_grid(128, 128), 256, 0, stream>>>(PW(base + 0), wpk, 128, 128, 128);
    gemm_edge_wmma<64, 128><<<gemm_grid(M_EDGE, 128), 256, 0, stream>>>(
        fcur, idx, wpk, PW(base + 2), PW(base + 1), PW(base + 3), PW(base + 4), cvA);

    pack_w_kernel<<<pack_grid(128, 128), 256, 0, stream>>>(PW(base + 5), wpk, 128, 128, 128);
    gemm_f16_wmma<128, 128><<<gemm_grid(M_EDGE, 128), 256, 0, stream>>>(
        cvA, wpk, PW(base + 7), PW(base + 6), PW(base + 8), PW(base + 9), cvB);

    pack_w_kernel<<<pack_grid(128, 128), 256, 0, stream>>>(PW(base + 10), wpk, 128, 128, 128);
    gemm_f16_wmma<128, 128><<<gemm_grid(M_EDGE, 128), 256, 0, stream>>>(
        cvB, wpk, PW(base + 12), PW(base + 11), PW(base + 13), PW(base + 14), cvA);

    maxk_kernel<<<(NP * 128 + 255) / 256, 256, 0, stream>>>(cvA, agg, 128);

    pack_w_kernel<<<pack_grid(64, 128), 256, 0, stream>>>(PW(base + 15), wpk, 64, 64, 128);
    gemm_f32_wmma<64, 128, false, true><<<gemm_grid(NP, 128), 256, 0, stream>>>(
        fcur, wpk, nullptr, PW(base + 17), PW(base + 16), PW(base + 18), PW(base + 19),
        agg, fnxt, NP);
    float* tmp = fcur; fcur = fnxt; fnxt = tmp;
  }

  // ---------------- block 2: cin=128, ch=256 ----------------
  {
    const int base = 40;
    knn_topk_kernel<128><<<(NP + 63) / 64, 64, 0, stream>>>(fcur, idx);

    pack_w_kernel<<<pack_grid(256, 256), 256, 0, stream>>>(PW(base + 0), wpk, 256, 256, 256);
    gemm_edge_wmma<128, 256><<<gemm_grid(M_EDGE, 256), 256, 0, stream>>>(
        fcur, idx, wpk, PW(base + 2), PW(base + 1), PW(base + 3), PW(base + 4), cvA);

    pack_w_kernel<<<pack_grid(256, 256), 256, 0, stream>>>(PW(base + 5), wpk, 256, 256, 256);
    gemm_f16_wmma<256, 256><<<gemm_grid(M_EDGE, 256), 256, 0, stream>>>(
        cvA, wpk, PW(base + 7), PW(base + 6), PW(base + 8), PW(base + 9), cvB);

    pack_w_kernel<<<pack_grid(256, 256), 256, 0, stream>>>(PW(base + 10), wpk, 256, 256, 256);
    gemm_f16_wmma<256, 256><<<gemm_grid(M_EDGE, 256), 256, 0, stream>>>(
        cvB, wpk, PW(base + 12), PW(base + 11), PW(base + 13), PW(base + 14), cvA);

    maxk_kernel<<<(NP * 256 + 255) / 256, 256, 0, stream>>>(cvA, agg, 256);

    pack_w_kernel<<<pack_grid(128, 256), 256, 0, stream>>>(PW(base + 15), wpk, 128, 128, 256);
    gemm_f32_wmma<128, 256, false, true><<<gemm_grid(NP, 256), 256, 0, stream>>>(
        fcur, wpk, nullptr, PW(base + 17), PW(base + 16), PW(base + 18), PW(base + 19),
        agg, fnxt, NP);
    float* tmp = fcur; fcur = fnxt; fnxt = tmp;
  }

  // pool + VAE head
  pool_kernel<<<(N_B * 256 + 255) / 256, 256, 0, stream>>>(fcur, poolb, 256);
  encoder_kernel<<<(N_B * LATENT + 255) / 256, 256, 0, stream>>>(
      poolb, PW(I_ENC_WM), PW(I_ENC_WV), PW(I_ENC_BM), PW(I_ENC_BV), eps, z, zm, zv, 256);

  // decoder: dec0 = relu(bn(z @ W0 + b0)) -> (128, 6400) == (NP, 64)
  pack_w_kernel<<<pack_grid(32, 6400), 256, 0, stream>>>(PW(I_DEC_W0), wpk, 16, 32, 6400);
  gemm_f32_wmma<16, 6400, true, false><<<gemm_grid(N_B, 6400), 256, 0, stream>>>(
      z, wpk, PW(I_DEC_B0),
      PW(I_DEC_BN0 + 1), PW(I_DEC_BN0 + 0), PW(I_DEC_BN0 + 2), PW(I_DEC_BN0 + 3),
      nullptr, dxA, N_B);

  // dec conv 64 -> 128
  pack_w_kernel<<<pack_grid(64, 128), 256, 0, stream>>>(PW(I_DEC_CONV + 0), wpk, 64, 64, 128);
  gemm_f32_wmma<64, 128, false, false><<<gemm_grid(NP, 128), 256, 0, stream>>>(
      dxA, wpk, nullptr,
      PW(I_DEC_CONV + 2), PW(I_DEC_CONV + 1), PW(I_DEC_CONV + 3), PW(I_DEC_CONV + 4),
      nullptr, dxB, NP);

  // dec conv 128 -> 64
  pack_w_kernel<<<pack_grid(128, 64), 256, 0, stream>>>(PW(I_DEC_CONV + 5), wpk, 128, 128, 64);
  gemm_f32_wmma<128, 64, false, false><<<gemm_grid(NP, 64), 256, 0, stream>>>(
      dxB, wpk, nullptr,
      PW(I_DEC_CONV + 7), PW(I_DEC_CONV + 6), PW(I_DEC_CONV + 8), PW(I_DEC_CONV + 9),
      nullptr, dxA, NP);

  // output head
  dec_out_kernel<<<(NP * 3 + 255) / 256, 256, 0, stream>>>(
      dxA, PW(I_DEC_WOUT), PW(I_DEC_BOUT), dec);
}
